// LengthRegulator_25185688224629
// MI455X (gfx1250) — matrix operand order repaired and checked by reading, not compiled, not run on hardware
//
#include <hip/hip_runtime.h>
#include <hip/hip_bf16.h>
#include <math.h>

typedef __attribute__((ext_vector_type(16))) __bf16 v16bf;
typedef __attribute__((ext_vector_type(8)))  __bf16 v8bf;
typedef __attribute__((ext_vector_type(8)))  float  v8f;
typedef __attribute__((ext_vector_type(4)))  float  v4f;   // clang vector: OK for NT builtins

#define TENC 512
#define DCH  256
#define TMEL 4096
#define NB   16
#define KTOT 768            // 3 * 256
#define KSTEPS 24           // 768 / 32
#define ASTRIDE 264         // padded bf16 row stride for A tile in LDS

// ---------------------------------------------------------------------------
// Pre-pack conv weights (f32 [3,Cin,Cout] row-major) into bf16 WMMA B-fragment
// order: wp[ntile][kstep][lane][e], so a wave's B fragment for one K-step is a
// contiguous, coalesced 1 KiB block (two global_load_b128 per lane).
//   lane: N = ntile*16 + (lane&15); elements e: K = kstep*32 + (lane>>4)*16 + e
// ---------------------------------------------------------------------------
__global__ __launch_bounds__(256) void pack_weights_kernel(
    const float* __restrict__ w,   // [768*256]
    __bf16* __restrict__ wp)       // [16*24*32*16]
{
    const int blk = blockIdx.x;            // nt*24 + ks, 0..383
    const int ks  = blk % KSTEPS;
    const int nt  = blk / KSTEPS;
    const int t    = threadIdx.x;          // 256 threads -> 512 values
    const int lane = t >> 3;               // 0..31
    const int e0   = (t & 7) * 2;
    #pragma unroll
    for (int q = 0; q < 2; ++q) {
        const int e  = e0 + q;
        const int kv = ks * 32 + (lane >> 4) * 16 + e;
        const int n  = nt * 16 + (lane & 15);
        wp[((long)(blk * 32 + lane) * 16) + e] = (__bf16)w[(long)kv * DCH + n];
    }
}

// ---------------------------------------------------------------------------
// Fused conv1d(K=3,SAME) + bias + LayerNorm + ReLU.
// Block (256 thr, 8 waves) computes 16 rows x 256 cols. A tile (18 rows) is
// staged once into LDS as bf16 (uniform boundary branches); each wave runs two
// 16x16 tiles of 24 v_wmma_f32_16x16x32_bf16, fragments via ds_load_b128 (A)
// and coalesced global_load_b128 (pre-packed B).
// ---------------------------------------------------------------------------
__global__ __launch_bounds__(256) void conv_ln_relu_kernel(
    const float*  __restrict__ in,    // [B*TENC, 256] f32
    const __bf16* __restrict__ wp,    // packed weights
    const float*  __restrict__ bias,
    const float*  __restrict__ gamma,
    const float*  __restrict__ beta,
    float* __restrict__ out)          // [B*TENC, 256] f32
{
    __shared__ __align__(16) __bf16 Albs[18 * ASTRIDE];
    __shared__ float lds_y[16 * 257];          // (r*257+c) % 64 conflict-free
    __shared__ float psum[16][16];
    __shared__ float psq[16][16];
    __shared__ float lds_mu[16];
    __shared__ float lds_rs[16];

    const int row0 = blockIdx.x * 16;          // global row = b*512 + t
    const int b    = row0 / TENC;
    const int t0   = row0 % TENC;              // 512 % 16 == 0

    const int tid   = threadIdx.x;
    const int wave  = tid >> 5;
    const int lane  = tid & 31;
    const int mlane = lane & 15;
    const int hi    = lane >> 4;

    // ---- stage A tile: rows t0-1 .. t0+16, bf16, boundary rows zeroed ----
    #pragma unroll 3
    for (int r = 0; r < 18; ++r) {
        const int trow = t0 + r - 1;           // uniform per iteration
        float v = 0.0f;
        if (trow >= 0 && trow < TENC)
            v = in[((long)b * TENC + trow) * DCH + tid];
        Albs[r * ASTRIDE + tid] = (__bf16)v;
    }
    __syncthreads();

    #pragma unroll
    for (int nt = 0; nt < 2; ++nt) {
        const int ntile = wave + nt * 8;
        const int ncol  = ntile * 16 + mlane;
        const __bf16* bsrc = wp + ((long)ntile * KSTEPS * 32 + lane) * 16;

        v8f acc = {0.f, 0.f, 0.f, 0.f, 0.f, 0.f, 0.f, 0.f};

        #pragma unroll 4
        for (int ks = 0; ks < KSTEPS; ++ks) {
            const int kk  = ks * 32;
            // A fragment: element e -> K = kk + hi*16 + e
            const int kb0 = kk + hi * 16;
            const int kb1 = kk + hi * 16 + 8;
            const v8bf a0 = *(const v8bf*)&Albs[(mlane + (kb0 >> 8)) * ASTRIDE + (kb0 & 255)];
            const v8bf a1 = *(const v8bf*)&Albs[(mlane + (kb1 >> 8)) * ASTRIDE + (kb1 & 255)];
            const v16bf afrag = __builtin_shufflevector(a0, a1,
                0, 1, 2, 3, 4, 5, 6, 7, 8, 9, 10, 11, 12, 13, 14, 15);
            // B fragment: contiguous 32 B per lane from packed weights
            const __bf16* bp = bsrc + (long)ks * 512;
            const v8bf b0 = *(const v8bf*)bp;
            const v8bf b1 = *(const v8bf*)(bp + 8);
            const v16bf bfrag = __builtin_shufflevector(b0, b1,
                0, 1, 2, 3, 4, 5, 6, 7, 8, 9, 10, 11, 12, 13, 14, 15);

            acc = __builtin_amdgcn_wmma_f32_16x16x32_bf16(
                false, afrag, false, bfrag, (short)0, acc, false, false);
        }

        // C/D layout: VGPR r -> (M = r + hi*8, N = ncol)
        const float bn = bias[ncol];
        #pragma unroll
        for (int r = 0; r < 8; ++r)
            lds_y[(r + hi * 8) * 257 + ncol] = acc[r] + bn;
    }
    __syncthreads();

    // ---- LayerNorm: two-level reduction over 256 cols per row ----
    {
        const int m  = tid >> 4;
        const int c0 = (tid & 15) * 16;
        float s = 0.f, s2 = 0.f;
        #pragma unroll
        for (int c = 0; c < 16; ++c) {
            const float v = lds_y[m * 257 + c0 + c];
            s += v; s2 += v * v;
        }
        psum[m][tid & 15] = s;
        psq [m][tid & 15] = s2;
    }
    __syncthreads();
    if (tid < 16) {
        float s = 0.f, s2 = 0.f;
        #pragma unroll
        for (int k = 0; k < 16; ++k) { s += psum[tid][k]; s2 += psq[tid][k]; }
        const float mu  = s * (1.0f / DCH);
        const float var = s2 * (1.0f / DCH) - mu * mu;
        lds_mu[tid] = mu;
        lds_rs[tid] = rsqrtf(var + 1e-5f);
    }
    __syncthreads();

    // ---- scale/shift + ReLU, coalesced writeback ----
    {
        const int m  = tid >> 4;
        const int c0 = (tid & 15) * 16;
        const float mu = lds_mu[m];
        const float rs = lds_rs[m];
        float* dst = out + (long)(row0 + m) * DCH + c0;
        #pragma unroll
        for (int c = 0; c < 16; ++c) {
            const int cc = c0 + c;
            const float v = (lds_y[m * 257 + cc] - mu) * rs * gamma[cc] + beta[cc];
            dst[c] = fmaxf(v, 0.0f);
        }
    }
}

// ---------------------------------------------------------------------------
// dpo[row] = exp(h[row,:] . wl + bl) ; one wave32 per row
// ---------------------------------------------------------------------------
__global__ __launch_bounds__(256) void linear_exp_kernel(
    const float* __restrict__ h, const float* __restrict__ wl,
    const float* __restrict__ bl, float* __restrict__ dpo)
{
    const int row  = blockIdx.x * 8 + (threadIdx.x >> 5);
    const int lane = threadIdx.x & 31;
    const float* src = h + (long)row * DCH;
    float s = 0.f;
    #pragma unroll
    for (int c = lane; c < DCH; c += 32) s += src[c] * wl[c];
    #pragma unroll
    for (int off = 16; off > 0; off >>= 1) s += __shfl_down(s, off, 32);
    if (lane == 0) dpo[row] = expf(s + bl[0]);
}

// ---------------------------------------------------------------------------
// Per-batch inclusive cumsum of target durations
// ---------------------------------------------------------------------------
__global__ void cumsum_kernel(const int* __restrict__ target, int* __restrict__ cum)
{
    const int b = threadIdx.x;
    if (b < NB) {
        int s = 0;
        const int* tp = target + b * TENC;
        int* cp = cum + b * TENC;
        for (int t = 0; t < TENC; ++t) { s += tp[t]; cp[t] = s; }
    }
}

// ---------------------------------------------------------------------------
// Length regulator: one wave32 per (b, m) row; binary search + gather.
// 192 MiB of write-once output -> nontemporal 16B vector stores (TH=NT),
// keeping L2 free for the x gather.
// ---------------------------------------------------------------------------
__global__ __launch_bounds__(256) void length_regulate_kernel(
    const float* __restrict__ x,    // [16, 512, 256]
    const int* __restrict__ cum,    // [16, 512]
    float* __restrict__ outp,       // [16, TMEL, 256]
    float* __restrict__ align)      // [16, TMEL, 512]
{
    const long gid  = (long)blockIdx.x * 8 + (threadIdx.x >> 5);
    const int  lane = threadIdx.x & 31;
    const int  b    = (int)(gid >> 12);
    const int  m    = (int)(gid & 4095);

    const int* cb = cum + b * TENC;
    int j = -1;
    if (m < cb[TENC - 1]) {
        int lo = 0, hh = TENC - 1;          // invariant: cb[hh] > m
        while (lo < hh) {
            const int mid = (lo + hh) >> 1;
            if (cb[mid] > m) hh = mid; else lo = mid + 1;
        }
        j = lo;
    }

    // alignment row: 512 floats; lane covers cols [lane*16, lane*16+16)
    {
        v4f* arow = (v4f*)(align + ((long)b * TMEL + m) * TENC) + lane * 4;
        #pragma unroll
        for (int q = 0; q < 4; ++q) {
            v4f v = {0.f, 0.f, 0.f, 0.f};
            const int cbase = lane * 16 + q * 4;
            if (j >= cbase && j < cbase + 4)
                v[j - cbase] = 1.0f;
            __builtin_nontemporal_store(v, arow + q);
        }
    }
    // output row: 256 floats; lane writes 16B-vector indices {lane, lane+32}
    {
        v4f* orow = (v4f*)(outp + ((long)b * TMEL + m) * DCH);
        if (j >= 0) {
            const v4f* xr = (const v4f*)(x + ((long)b * TENC + j) * DCH);
            __builtin_nontemporal_store(xr[lane],      orow + lane);
            __builtin_nontemporal_store(xr[lane + 32], orow + lane + 32);
        } else {
            const v4f z = {0.f, 0.f, 0.f, 0.f};
            __builtin_nontemporal_store(z, orow + lane);
            __builtin_nontemporal_store(z, orow + lane + 32);
        }
    }
}

// ---------------------------------------------------------------------------
extern "C" void kernel_launch(void* const* d_in, const int* in_sizes, int n_in,
                              void* d_out, int out_size, void* d_ws, size_t ws_size,
                              hipStream_t stream)
{
    const float* x      = (const float*)d_in[0];
    const int*   target = (const int*)  d_in[1];
    const float* w1     = (const float*)d_in[2];
    const float* b1     = (const float*)d_in[3];
    const float* g1     = (const float*)d_in[4];
    const float* be1    = (const float*)d_in[5];
    const float* w2     = (const float*)d_in[6];
    const float* b2     = (const float*)d_in[7];
    const float* g2     = (const float*)d_in[8];
    const float* be2    = (const float*)d_in[9];
    const float* wl     = (const float*)d_in[10];
    const float* bl     = (const float*)d_in[11];
    // d_in[12] = mel_max_length (4096, fixed by the reference shapes)

    float* outp  = (float*)d_out;                               // [16,4096,256]
    float* align = outp  + (size_t)NB * TMEL * DCH;             // [16,4096,512]
    float* dpo   = align + (size_t)NB * TMEL * TENC;            // [16,512]

    float*  h1   = (float*)d_ws;                                // 8 MiB
    float*  h2   = h1 + (size_t)NB * TENC * DCH;                // 8 MiB
    int*    cum  = (int*)(h2 + (size_t)NB * TENC * DCH);        // 32 KiB
    __bf16* wp1  = (__bf16*)(cum + NB * TENC);                  // 384 KiB
    __bf16* wp2  = wp1 + (size_t)KTOT * DCH;                    // 384 KiB

    const int mtiles = (NB * TENC) / 16;                        // 512

    pack_weights_kernel<<<16 * KSTEPS, 256, 0, stream>>>(w1, wp1);
    pack_weights_kernel<<<16 * KSTEPS, 256, 0, stream>>>(w2, wp2);
    conv_ln_relu_kernel<<<mtiles, 256, 0, stream>>>(x,  wp1, b1, g1, be1, h1);
    conv_ln_relu_kernel<<<mtiles, 256, 0, stream>>>(h1, wp2, b2, g2, be2, h2);
    linear_exp_kernel  <<<(NB * TENC) / 8, 256, 0, stream>>>(h2, wl, bl, dpo);
    cumsum_kernel      <<<1, 32, 0, stream>>>(target, cum);
    length_regulate_kernel<<<(NB * TMEL) / 8, 256, 0, stream>>>(x, cum, outp, align);
}
